// BertSelfAttention_flashattn_71871982731347
// MI455X (gfx1250) — matrix operand order repaired and checked by reading, not compiled, not run on hardware
//
#include <hip/hip_runtime.h>
#include <hip/hip_bf16.h>

// ---------------------------------------------------------------------------
// BERT self-attention forward for MI455X (gfx1250, wave32, WMMA bf16).
//   qkv = x @ Wqkv^T + b ; flash-attn with key-padding mask ; out = ctx @ W^T + b
// All matmuls use v_wmma_f32_16x16x32_bf16 with fp32 accumulation.
// Tile staging uses CDNA5 async global->LDS copies (ASYNCcnt) with
// double-buffered LDS in the GEMMs.
// ---------------------------------------------------------------------------

typedef unsigned short u16;
typedef __attribute__((ext_vector_type(16))) __bf16         v16bf;
typedef __attribute__((ext_vector_type(8)))  float          v8f;
typedef __attribute__((ext_vector_type(4)))  float          v4f;
typedef __attribute__((ext_vector_type(4)))  unsigned int   v4u;
typedef __attribute__((ext_vector_type(4)))  unsigned short v4s;

#define B_ 8
#define S_ 1024
#define E_ 768
#define H_ 12
#define D_ 64

__device__ __forceinline__ u16 f2bf(float x) {
  unsigned u = __float_as_uint(x);
  u += 0x7FFFu + ((u >> 16) & 1u);   // round-to-nearest-even
  return (u16)(u >> 16);
}

union Frag32B { v4u q[2]; v16bf v; };

// Load a 32-byte WMMA operand fragment as two 16-byte LDS reads (ds_load_b128).
__device__ __forceinline__ v16bf ld_frag(const u16* p0, const u16* p1) {
  Frag32B f;
  f.q[0] = *(const v4u*)p0;
  f.q[1] = *(const v4u*)p1;
  return f.v;
}

// CDNA5 async global->LDS copy, 16 bytes per lane (GLOBAL_LOAD_ASYNC_TO_LDS_B128,
// tracked by ASYNCcnt). LDS flat addresses carry the wave-relative LDS byte
// offset in their low 32 bits (ISA 10.2 aperture rules), which is what the
// VDST register of the async op expects.
__device__ __forceinline__ void async_g2l_b128(const void* gsrc, void* ldst) {
  unsigned loff = (unsigned)(unsigned long long)ldst;
  asm volatile("global_load_async_to_lds_b128 %0, %1, off"
               :: "v"(loff), "v"(gsrc)
               : "memory");
}

template <int N>
__device__ __forceinline__ void wait_asynccnt() {
  asm volatile("s_wait_asynccnt %0" :: "i"(N) : "memory");
}

// ---------------------------------------------------------------------------
// fp32 -> bf16 conversion (vectorized x4)
// ---------------------------------------------------------------------------
__global__ void cvt_f32_to_bf16(const float* __restrict__ src,
                                u16* __restrict__ dst, int n4) {
  int i = blockIdx.x * blockDim.x + threadIdx.x;
  if (i >= n4) return;
  v4f f = ((const v4f*)src)[i];
  v4s o;
  o[0] = f2bf(f[0]); o[1] = f2bf(f[1]); o[2] = f2bf(f[2]); o[3] = f2bf(f[3]);
  ((v4s*)dst)[i] = o;
}

// ---------------------------------------------------------------------------
// C[M,N] = A[M,K] @ W[N,K]^T + bias[N]
// Block tile 128x128, BK=32, 256 threads = 8 waves (2x4), wave tile 64x32.
// Double-buffered LDS fed by async global->LDS copies:
//   issue tile it+1  ->  s_wait_asynccnt 4 (tile it done, in-order)  ->
//   barrier  ->  8 WMMAs on tile it  ->  barrier (protects buffer reuse).
// A-frag: lane m=lane&15, half=lane>>4; a[0..7]=K half*8+0..7, a[8..15]=+16.
// B-frag (row of W = column of B): b[0..7]=K half*8+0..7, b[8..15]=+16.
// ---------------------------------------------------------------------------
template <bool STORE_BF16>
__global__ __launch_bounds__(256)
void gemm_xwT_wmma(const u16* __restrict__ A, const u16* __restrict__ W,
                   const float* __restrict__ bias, void* __restrict__ outp,
                   int M, int N, int K) {
  __shared__ u16 As[2][128 * 32];
  __shared__ u16 Ws[2][128 * 32];

  const int t    = threadIdx.x;
  const int lane = t & 31;
  const int wave = t >> 5;
  const int wm   = (wave >> 2) * 64;   // wave M offset in block
  const int wn   = (wave & 3) * 32;    // wave N offset in block
  const int mlo  = lane & 15;
  const int hl   = lane >> 4;
  const int bm   = blockIdx.y * 128;
  const int bn   = blockIdx.x * 128;

  // This thread's slice of the cooperative tile copy (4 async ops / tile).
  const int r0 = t >> 2;                 // rows for j=0 slice
  const int c0 = (t & 3) * 8;
  const int r1 = (t + 256) >> 2;         // rows for j=1 slice
  const int c1 = c0;

  v8f acc[4][2];
#pragma unroll
  for (int mi = 0; mi < 4; ++mi) {
#pragma unroll
    for (int ni = 0; ni < 2; ++ni) acc[mi][ni] = (v8f)0.0f;
  }

  auto issue_tile = [&](int buf, int k0) {
    async_g2l_b128(&A[(size_t)(bm + r0) * K + k0 + c0], &As[buf][r0 * 32 + c0]);
    async_g2l_b128(&W[(size_t)(bn + r0) * K + k0 + c0], &Ws[buf][r0 * 32 + c0]);
    async_g2l_b128(&A[(size_t)(bm + r1) * K + k0 + c1], &As[buf][r1 * 32 + c1]);
    async_g2l_b128(&W[(size_t)(bn + r1) * K + k0 + c1], &Ws[buf][r1 * 32 + c1]);
  };

  const int nk = K / 32;
  issue_tile(0, 0);

  for (int it = 0; it < nk; ++it) {
    const int cur = it & 1;
    if (it + 1 < nk) {
      issue_tile(cur ^ 1, (it + 1) * 32);
      wait_asynccnt<4>();   // tile `it` complete (async loads finish in order)
    } else {
      wait_asynccnt<0>();
    }
    __syncthreads();        // every wave's copies for tile `it` have landed

    v16bf af[4], wf[2];
#pragma unroll
    for (int mi = 0; mi < 4; ++mi) {
      const u16* p = &As[cur][(wm + mi * 16 + mlo) * 32];
      af[mi] = ld_frag(p + hl * 8, p + 16 + hl * 8);
    }
#pragma unroll
    for (int ni = 0; ni < 2; ++ni) {
      const u16* p = &Ws[cur][(wn + ni * 16 + mlo) * 32];
      wf[ni] = ld_frag(p + hl * 8, p + 16 + hl * 8);
    }
#pragma unroll
    for (int mi = 0; mi < 4; ++mi) {
#pragma unroll
      for (int ni = 0; ni < 2; ++ni) {
        acc[mi][ni] = __builtin_amdgcn_wmma_f32_16x16x32_bf16(
            false, af[mi], false, wf[ni], (short)0, acc[mi][ni], false, false);
      }
    }
    __syncthreads();        // done reading buf `cur` before it+1 refills it
  }

  // Epilogue: C layout — VGPR v: lanes 0-15 -> M=v, lanes 16-31 -> M=v+8.
#pragma unroll
  for (int mi = 0; mi < 4; ++mi) {
#pragma unroll
    for (int ni = 0; ni < 2; ++ni) {
#pragma unroll
      for (int v = 0; v < 8; ++v) {
        int row   = bm + wm + mi * 16 + v + hl * 8;
        int col   = bn + wn + ni * 16 + mlo;
        float val = acc[mi][ni][v] + bias[col];
        if (STORE_BF16)
          ((u16*)outp)[(size_t)row * N + col] = f2bf(val);
        else
          ((float*)outp)[(size_t)row * N + col] = val;
      }
    }
  }
}

// ---------------------------------------------------------------------------
// Flash attention: one block per (b*H + h, 128-query tile); 8 waves x 16 rows.
// qkv layout: [B*S, 3*E] bf16; q at +0, k at +E, v at +2E, head slice h*D.
// Q and K tiles staged via async global->LDS; V needs a transposed LDS layout
// so it goes through VGPRs.
// ---------------------------------------------------------------------------
__global__ __launch_bounds__(256)
void flash_attn_wmma(const u16* __restrict__ qkv,
                     const unsigned char* __restrict__ kpm,  // jnp bool_: 1B/elem
                     u16* __restrict__ ctx) {
  __shared__ u16   Qs[128 * 64];      // 16 KB  query tile, row-major [m][d]
  __shared__ u16   Ks[64 * 64];       //  8 KB  key tile,   row-major [key][d]
  __shared__ u16   Vts[64 * 64];      //  8 KB  value tile, transposed [d][key]
  __shared__ u16   Ps[8 * 16 * 64];   // 16 KB  per-wave P pane (C->A relayout)
  __shared__ float maskb[64];         // additive key mask bias
  __shared__ float qmaskf[128];       // query mask (0/1)

  const int t = threadIdx.x, lane = t & 31, wave = t >> 5;
  const int mlo = lane & 15, hl = lane >> 4;
  const int b  = blockIdx.x / H_;
  const int h  = blockIdx.x % H_;
  const int q0 = blockIdx.y * 128;
  const int rs3 = 3 * E_;             // 2304

  // Stage Q tile (async) + query mask.
#pragma unroll
  for (int j = 0; j < 4; ++j) {
    int i = t + 256 * j;
    int r = i >> 3;
    int c = (i & 7) * 8;
    async_g2l_b128(&qkv[(size_t)(b * S_ + q0 + r) * rs3 + h * D_ + c],
                   &Qs[r * 64 + c]);
  }
  if (t < 128) qmaskf[t] = kpm[b * S_ + q0 + t] ? 1.0f : 0.0f;
  wait_asynccnt<0>();
  __syncthreads();

  // Q A-fragments (reused for all key tiles): 2 k-steps over D=64.
  v16bf qa[2];
  {
    const u16* p = &Qs[(wave * 16 + mlo) * 64];
#pragma unroll
    for (int ks = 0; ks < 2; ++ks)
      qa[ks] = ld_frag(p + ks * 32 + hl * 8, p + ks * 32 + 16 + hl * 8);
  }

  v8f o[4];
#pragma unroll
  for (int ni = 0; ni < 4; ++ni) o[ni] = (v8f)0.0f;
  float mrow[8], lrow[8];
#pragma unroll
  for (int v = 0; v < 8; ++v) { mrow[v] = -3.0e38f; lrow[v] = 0.0f; }

  u16* myP = &Ps[wave * (16 * 64)];

  for (int kt = 0; kt < S_ / 64; ++kt) {
    __syncthreads();  // previous iteration done reading Ks/Vts
#pragma unroll
    for (int j = 0; j < 2; ++j) {
      int i = t + 256 * j;
      int r = i >> 3;
      int c = (i & 7) * 8;
      size_t base = (size_t)(b * S_ + kt * 64 + r) * rs3 + h * D_ + c;
      async_g2l_b128(&qkv[base + E_], &Ks[r * 64 + c]);   // K: direct copy
      v4u vv = *(const v4u*)&qkv[base + 2 * E_];          // V[key=r][d=c..c+7]
      const u16* ve = (const u16*)&vv;
#pragma unroll
      for (int e = 0; e < 8; ++e) Vts[(c + e) * 64 + r] = ve[e];  // transpose
    }
    if (t < 64) maskb[t] = kpm[b * S_ + kt * 64 + t] ? 0.0f : -1.0e30f;
    wait_asynccnt<0>();
    __syncthreads();

    // S = (Q K^T) * 1/sqrt(D) + mask    (B-frag = K tile row, d-contiguous)
    v8f sf[4];
#pragma unroll
    for (int ni = 0; ni < 4; ++ni) {
      sf[ni] = (v8f)0.0f;
#pragma unroll
      for (int ks = 0; ks < 2; ++ks) {
        const u16* p = &Ks[(ni * 16 + mlo) * 64 + ks * 32 + hl * 8];
        v16bf kb = ld_frag(p, p + 16);
        sf[ni] = __builtin_amdgcn_wmma_f32_16x16x32_bf16(
            false, qa[ks], false, kb, (short)0, sf[ni], false, false);
      }
      float mb = maskb[ni * 16 + mlo];  // lane's 8 values share column N
#pragma unroll
      for (int v = 0; v < 8; ++v) sf[ni][v] = sf[ni][v] * 0.125f + mb;
    }

    // Online softmax. Row M=v+hl*8 lives across the 16-lane half-wave.
#pragma unroll
    for (int v = 0; v < 8; ++v) {
      float x = fmaxf(fmaxf(sf[0][v], sf[1][v]), fmaxf(sf[2][v], sf[3][v]));
#pragma unroll
      for (int msk = 8; msk >= 1; msk >>= 1)
        x = fmaxf(x, __shfl_xor(x, msk, 32));
      float mnew  = fmaxf(mrow[v], x);
      float alpha = __expf(mrow[v] - mnew);
      mrow[v]     = mnew;
      float rsum  = 0.0f;
#pragma unroll
      for (int ni = 0; ni < 4; ++ni) {
        float e0 = __expf(sf[ni][v] - mnew);
        sf[ni][v] = e0;
        rsum += e0;
      }
#pragma unroll
      for (int msk = 8; msk >= 1; msk >>= 1)
        rsum += __shfl_xor(rsum, msk, 32);
      lrow[v] = lrow[v] * alpha + rsum;
#pragma unroll
      for (int ni = 0; ni < 4; ++ni) o[ni][v] *= alpha;
    }

    // Spill P (C layout) to per-wave LDS pane as bf16, reload in A layout.
#pragma unroll
    for (int ni = 0; ni < 4; ++ni) {
#pragma unroll
      for (int v = 0; v < 8; ++v)
        myP[(v + hl * 8) * 64 + ni * 16 + mlo] = f2bf(sf[ni][v]);
    }
    // DS ops are wave-in-order; wait pins compiler ordering of the reload.
    asm volatile("s_wait_dscnt 0" ::: "memory");

    v16bf pa[2];
    {
      const u16* p = &myP[mlo * 64];
#pragma unroll
      for (int ks = 0; ks < 2; ++ks)
        pa[ks] = ld_frag(p + ks * 32 + hl * 8, p + ks * 32 + 16 + hl * 8);
    }
    // O += P @ V   (B-frag = Vts row d, key-contiguous)
#pragma unroll
    for (int ni = 0; ni < 4; ++ni) {
#pragma unroll
      for (int ks = 0; ks < 2; ++ks) {
        const u16* p = &Vts[(ni * 16 + mlo) * 64 + ks * 32 + hl * 8];
        v16bf vb = ld_frag(p, p + 16);
        o[ni] = __builtin_amdgcn_wmma_f32_16x16x32_bf16(
            false, pa[ks], false, vb, (short)0, o[ni], false, false);
      }
    }
  }

  // Normalize by 1/l, apply query mask, store ctx as bf16 [B*S, E].
#pragma unroll
  for (int v = 0; v < 8; ++v) {
    int qrow  = q0 + wave * 16 + v + hl * 8;
    float inv = qmaskf[wave * 16 + v + hl * 8] / lrow[v];
#pragma unroll
    for (int ni = 0; ni < 4; ++ni) {
      float val = o[ni][v] * inv;
      ctx[(size_t)(b * S_ + qrow) * E_ + h * D_ + ni * 16 + mlo] = f2bf(val);
    }
  }
}

// ---------------------------------------------------------------------------
// Workspace layout (bytes, 256-aligned):
//   xb    @ 0         : 8192*768*2   = 12,582,912
//   wqkvb @ 12582912  : 2304*768*2   =  3,538,944
//   owb   @ 16121856  :  768*768*2   =  1,179,648
//   qkvb  @ 17301504  : 8192*2304*2  = 37,748,736
//   ctxb  @ 55050240  : 8192*768*2   = 12,582,912   (total ~67.6 MB)
// ---------------------------------------------------------------------------
extern "C" void kernel_launch(void* const* d_in, const int* in_sizes, int n_in,
                              void* d_out, int out_size, void* d_ws,
                              size_t ws_size, hipStream_t stream) {
  (void)in_sizes; (void)n_in; (void)out_size; (void)ws_size;
  const float* x             = (const float*)d_in[0];
  const unsigned char* kpm   = (const unsigned char*)d_in[1];
  const float* wqkv          = (const float*)d_in[2];
  const float* bqkv          = (const float*)d_in[3];
  const float* ow            = (const float*)d_in[4];
  const float* ob            = (const float*)d_in[5];
  float* out                 = (float*)d_out;

  char* ws   = (char*)d_ws;
  u16* xb    = (u16*)(ws + 0);
  u16* wqkvb = (u16*)(ws + 12582912);
  u16* owb   = (u16*)(ws + 16121856);
  u16* qkvb  = (u16*)(ws + 17301504);
  u16* ctxb  = (u16*)(ws + 55050240);

  cvt_f32_to_bf16<<<(8192 * 768 / 4 + 255) / 256, 256, 0, stream>>>(
      x, xb, 8192 * 768 / 4);
  cvt_f32_to_bf16<<<(2304 * 768 / 4 + 255) / 256, 256, 0, stream>>>(
      wqkv, wqkvb, 2304 * 768 / 4);
  cvt_f32_to_bf16<<<(768 * 768 / 4 + 255) / 256, 256, 0, stream>>>(
      ow, owb, 768 * 768 / 4);

  // qkv = x @ Wqkv^T + b   (bf16 out for attention consumption)
  gemm_xwT_wmma<true><<<dim3(2304 / 128, 8192 / 128), 256, 0, stream>>>(
      xb, wqkvb, bqkv, (void*)qkvb, 8192, 2304, 768);

  // flash attention over 96 (b,h) pairs x 8 query tiles
  flash_attn_wmma<<<dim3(B_ * H_, S_ / 128), 256, 0, stream>>>(qkvb, kpm, ctxb);

  // out = ctx @ out_w^T + b  (fp32 out)
  gemm_xwT_wmma<false><<<dim3(768 / 128, 8192 / 128), 256, 0, stream>>>(
      ctxb, owb, ob, (void*)out, 8192, 768, 768);
}
